// GloVe_42872363549012
// MI455X (gfx1250) — matrix operand order repaired and checked by reading, not compiled, not run on hardware
//
#include <hip/hip_runtime.h>
#include <hip/hip_bf16.h>

#define VOCAB 8192
#define DIM   256
#define TILE  128
#define KCHUNK 64
#define PITCH  72   // bf16 elems per LDS row: 64 + 8 pad -> 36-dword pitch, conflict-free

typedef __attribute__((ext_vector_type(16))) __bf16 v16bf;
typedef __attribute__((ext_vector_type(8)))  float  v8f;

__device__ __forceinline__ unsigned short f2bf(float f) {
    // round-to-nearest-even f32 -> bf16
    unsigned int u = __builtin_bit_cast(unsigned int, f);
    u += 0x7FFFu + ((u >> 16) & 1u);
    return (unsigned short)(u >> 16);
}

union Frag {
    struct { uint4 lo, hi; } q;   // two 16-byte LDS segments
    v16bf v;                      // 16 bf16 lanes-worth for WMMA
};

__global__ __launch_bounds__(256) void glove_tile_kernel(
    const float* __restrict__ Xw, const float* __restrict__ X,
    const float* __restrict__ W,  const float* __restrict__ WC,
    const float* __restrict__ b,  const float* __restrict__ bc,
    float* __restrict__ partials)
{
    __shared__ unsigned short Alds[TILE * PITCH];   // W tile, bf16
    __shared__ unsigned short Blds[TILE * PITCH];   // WC tile, bf16
    __shared__ float red[256];

    const int tid  = threadIdx.x;
    const int lane = tid & 31;
    const int wave = tid >> 5;            // 0..7, owns 16-row strip of the tile
    const int i0   = blockIdx.y * TILE;
    const int j0   = blockIdx.x * TILE;

    const int r    = tid >> 1;            // 0..127: row of tile this thread stages
    const int half = tid & 1;             // which 32-col half of the 64-col chunk
    const int g    = lane >> 4;           // lane group for fragment layout
    const int lr   = lane & 15;

    // ---- warm L2 for the epilogue's X / Xw tile (global_prefetch_b8) ----
    {
        const float* xrow  = X  + (size_t)(i0 + r) * VOCAB + j0 + half * 64;
        const float* xwrow = Xw + (size_t)(i0 + r) * VOCAB + j0 + half * 64;
        __builtin_prefetch(xrow,       0, 1);
        __builtin_prefetch(xrow  + 32, 0, 1);
        __builtin_prefetch(xwrow,      0, 1);
        __builtin_prefetch(xwrow + 32, 0, 1);
    }

    v8f acc[8];
    v8f zero = {};
#pragma unroll
    for (int jt = 0; jt < 8; ++jt) acc[jt] = zero;

    for (int kc = 0; kc < DIM; kc += KCHUNK) {
        // ---- cooperative stage: global f32 -> bf16 LDS (float4 loads, ushort4 stores) ----
        {
            const float* wsrc = W  + (size_t)(i0 + r) * DIM + kc + half * 32;
            const float* csrc = WC + (size_t)(j0 + r) * DIM + kc + half * 32;
            unsigned short* adst = &Alds[r * PITCH + half * 32];
            unsigned short* bdst = &Blds[r * PITCH + half * 32];
#pragma unroll
            for (int q = 0; q < 8; ++q) {
                float4 wa = ((const float4*)wsrc)[q];
                float4 ca = ((const float4*)csrc)[q];
                ushort4 wb; wb.x = f2bf(wa.x); wb.y = f2bf(wa.y); wb.z = f2bf(wa.z); wb.w = f2bf(wa.w);
                ushort4 cb; cb.x = f2bf(ca.x); cb.y = f2bf(ca.y); cb.z = f2bf(ca.z); cb.w = f2bf(ca.w);
                *(ushort4*)(adst + 4 * q) = wb;
                *(ushort4*)(bdst + 4 * q) = cb;
            }
        }
        __syncthreads();

        // ---- 16 WMMAs (2 K-steps x 8 column tiles), software-pipelined B frags ----
        // A fragment (16x32 bf16): lane group g holds contiguous K = 16g..16g+15
        // (VGPR0-7).  B fragment (32x16 bf16): lane = col,
        // group g holds contiguous K = 16g..16g+15.
        {
            const unsigned short* arow = &Alds[(wave * 16 + lr) * PITCH];
            Frag a[2];
#pragma unroll
            for (int ksi = 0; ksi < 2; ++ksi) {
                const unsigned short* abase = arow + ksi * 32;
                a[ksi].q.lo = *(const uint4*)(abase + g * 16);
                a[ksi].q.hi = *(const uint4*)(abase + g * 16 + 8);
            }

            Frag bf[2];
            {   // preload (ks=0, jt=0)
                const unsigned short* bbase = &Blds[lr * PITCH + g * 16];
                bf[0].q.lo = *(const uint4*)(bbase);
                bf[0].q.hi = *(const uint4*)(bbase + 8);
            }
#pragma unroll
            for (int t = 0; t < 16; ++t) {
                const int ksi = t >> 3;
                const int jt  = t & 7;
                const int cur = t & 1;
                if (t < 15) {   // issue next B-fragment load before this WMMA
                    const int nksi = (t + 1) >> 3;
                    const int njt  = (t + 1) & 7;
                    const unsigned short* bbase =
                        &Blds[(njt * 16 + lr) * PITCH + nksi * 32 + g * 16];
                    bf[cur ^ 1].q.lo = *(const uint4*)(bbase);
                    bf[cur ^ 1].q.hi = *(const uint4*)(bbase + 8);
                }
                acc[jt] = __builtin_amdgcn_wmma_f32_16x16x32_bf16(
                    false, a[ksi].v, false, bf[cur].v, (short)0, acc[jt], false, false);
            }
        }
        __syncthreads();
    }

    // ---- fused GloVe loss epilogue: D element (m,n) = (v + 8*(lane>>4), lane&15) ----
    float lsum = 0.0f;
    const int gi  = i0 + wave * 16 + 8 * g;
    const int gj0 = j0 + lr;
#pragma unroll
    for (int jt = 0; jt < 8; ++jt) {
        const int j = gj0 + jt * 16;
        const float bias = b[j] + bc[j];
#pragma unroll
        for (int v = 0; v < 8; ++v) {
            const size_t idx = (size_t)(gi + v) * VOCAB + j;
            const float score = acc[jt][v] + bias;
            const float d = score - __logf(X[idx]);
            lsum = fmaf(Xw[idx] * d, d, lsum);
        }
    }

    red[tid] = lsum;
    __syncthreads();
#pragma unroll
    for (int s = 128; s > 0; s >>= 1) {
        if (tid < s) red[tid] += red[tid + s];
        __syncthreads();
    }
    if (tid == 0) partials[blockIdx.y * gridDim.x + blockIdx.x] = red[0];
}

__global__ __launch_bounds__(256) void glove_reduce_kernel(
    const float* __restrict__ partials, float* __restrict__ out, int n)
{
    __shared__ float red[256];
    float s = 0.0f;
    for (int i = threadIdx.x; i < n; i += 256) s += partials[i];
    red[threadIdx.x] = s;
    __syncthreads();
#pragma unroll
    for (int k = 128; k > 0; k >>= 1) {
        if (threadIdx.x < k) red[threadIdx.x] += red[threadIdx.x + k];
        __syncthreads();
    }
    if (threadIdx.x == 0) out[0] = red[0];
}

extern "C" void kernel_launch(void* const* d_in, const int* in_sizes, int n_in,
                              void* d_out, int out_size, void* d_ws, size_t ws_size,
                              hipStream_t stream) {
    const float* Xw = (const float*)d_in[0];
    const float* X  = (const float*)d_in[1];
    const float* W  = (const float*)d_in[2];
    const float* WC = (const float*)d_in[3];
    const float* b  = (const float*)d_in[4];
    const float* bc = (const float*)d_in[5];
    float* partials = (float*)d_ws;   // (VOCAB/TILE)^2 = 4096 floats = 16 KB

    dim3 grid(VOCAB / TILE, VOCAB / TILE);
    glove_tile_kernel<<<grid, 256, 0, stream>>>(Xw, X, W, WC, b, bc, partials);
    glove_reduce_kernel<<<1, 256, 0, stream>>>(partials, (float*)d_out,
                                               (VOCAB / TILE) * (VOCAB / TILE));
}
